// SpikeToGraphGNO_11003706212823
// MI455X (gfx1250) — matrix operand order repaired and checked by reading, not compiled, not run on hardware
//
#include <hip/hip_runtime.h>
#include <hip/hip_bf16.h>
#include <math.h>

typedef __attribute__((ext_vector_type(16))) _Float16 v16h;
typedef __attribute__((ext_vector_type(8)))  _Float16 v8h;
typedef __attribute__((ext_vector_type(8)))  float    v8f;
typedef __attribute__((ext_vector_type(4)))  float    v4f;
typedef __attribute__((ext_vector_type(4)))  unsigned int u32x4;
typedef __attribute__((ext_vector_type(8)))  int      i32x8;
typedef __attribute__((ext_vector_type(4)))  int      i32x4;

namespace {

constexpr int Bb = 2, Nn = 2048, Tt = 256, Dd = 128, Kk = 16, Ll = 3;

// ---- WMMA helpers (CDNA5 v_wmma_f32_16x16x32_f16) -------------------------
// A fragment (16x32 f16, stride = 128 halves in source):
//   lanes 0-15  : row = lane,    halves 0..7 = K 0..7,  halves 8..15 = K 16..23
//   lanes 16-31 : row = lane-16, halves 0..7 = K 8..15, halves 8..15 = K 24..31
__device__ __forceinline__ v16h load_a_frag(const _Float16* base, int kbase, int lane) {
  const int row  = lane & 15;
  const int koff = (lane >> 4) << 3; // 0 or 8
  v8h lo = *(const v8h*)(base + row * Dd + kbase + koff);
  v8h hi = *(const v8h*)(base + row * Dd + kbase + 16 + koff);
  return __builtin_shufflevector(lo, hi, 0,1,2,3,4,5,6,7,8,9,10,11,12,13,14,15);
}

// B fragment (32x16 f16): lane l holds column n = nt*16 + (l&15); K order as A frag.
__device__ __forceinline__ v16h load_b_frag(const _Float16* w, int kbase, int nt, int lane) {
  const int n = nt * 16 + (lane & 15);
  const int koff = (lane >> 4) << 3;
  v16h r;
  #pragma unroll
  for (int s = 0; s < 16; ++s)
    r[s] = w[(kbase + koff + (s < 8 ? s : s + 8)) * Dd + n];
  return r;
}

__device__ __forceinline__ v8f wmma_f16(v16h a, v16h b, v8f c) {
  return __builtin_amdgcn_wmma_f32_16x16x32_f16(false, a, false, b, (short)0, c, false, false);
}

// ---- TDM: async-copy 32KB (16384 f16) contiguous global -> LDS ------------
// D# per CDNA5 ISA ch.8: group0 = {count/type/addr}, group1 = {dims/strides}.
// Issued by one wave; completion via TENSORcnt.
#if __has_builtin(__builtin_amdgcn_tensor_load_to_lds)
__device__ __forceinline__ void tdm_load_32kb(const _Float16* src, unsigned lds_off) {
  const unsigned long long ga = (unsigned long long)src;
  u32x4 g0 = { 1u,                                   // count=1, user mode
               lds_off,                               // lds_addr (bytes)
               (unsigned)(ga & 0xffffffffu),          // global_addr[31:0]
               ((unsigned)((ga >> 32) & 0x01ffffffu)) // global_addr[56:32]
                 | (2u << 30) };                      // type=2 ("image")
  i32x8 g1 = { (int)0x00010000,   // workgroup_mask=0, data_size=1 (2 bytes)
               (int)0x40000000,   // tensor_dim0[15:0]=16384 (bits 63:48)
               (int)(1 << 16),    // tensor_dim0[31:16]=0, tensor_dim1[15:0]=1
               (int)0x40000000,   // tensor_dim1[31:16]=0, tile_dim0=16384
               1,                 // tile_dim1=1, tile_dim2=0
               16384,             // tensor_dim0_stride[31:0]
               0, 0 };
  i32x4 z4 = { 0, 0, 0, 0 };
#if defined(__clang_major__) && (__clang_major__ >= 23)
  i32x8 z8 = { 0, 0, 0, 0, 0, 0, 0, 0 };
  __builtin_amdgcn_tensor_load_to_lds(g0, g1, z4, z4, z8, 0);
#else
  __builtin_amdgcn_tensor_load_to_lds(g0, g1, z4, z4, 0);
#endif
#if __has_builtin(__builtin_amdgcn_s_wait_tensorcnt)
  __builtin_amdgcn_s_wait_tensorcnt(0);
#else
  asm volatile("s_wait_tensorcnt 0x0" ::: "memory");
#endif
}
#endif

// Stage 16384 halves of pre-converted f16 weights into sW (at dynamic-LDS
// offset lds_off). TDM path when available, cooperative copy otherwise.
// Caller must __syncthreads() afterwards. blockDim.x == 256.
__device__ __forceinline__ void stage_weights(_Float16* sW, const _Float16* src,
                                              int tid, int wave, unsigned lds_off) {
#if __has_builtin(__builtin_amdgcn_tensor_load_to_lds)
  (void)tid;
  if (wave == 0) tdm_load_32kb(src, lds_off);
#else
  (void)wave; (void)lds_off;
  for (int e = tid; e < 2048; e += 256) ((v8h*)sW)[e] = ((const v8h*)src)[e];
#endif
}

// ===========================================================================
// Kernel 0: convert GEMM B-weights f32 -> f16 once into workspace.
// layout (halves): [0)       3*16384  : k1_w u_j slice (rows 130..257) per layer
//                  [49152)   3*16384  : k2_w per layer
//                  [98304)   16384    : W_logits
//                  [114688)  16384    : W_weight
// ===========================================================================
__global__ void k_convert(const float* __restrict__ k1w,
                          const float* __restrict__ k2w,
                          const float* __restrict__ Wl,
                          const float* __restrict__ Ww,
                          _Float16* __restrict__ dst) {
  const int e = blockIdx.x * 256 + threadIdx.x; // 0..131071
  float v;
  if (e < 49152) {
    const int l = e >> 14, r = e & 16383;
    v = k1w[l * 33024 + 16640 + r];             // 258*128=33024, 130*128=16640
  } else if (e < 98304) {
    const int t = e - 49152;
    v = k2w[(t >> 14) * 16384 + (t & 16383)];
  } else if (e < 114688) {
    v = Wl[e - 98304];
  } else {
    v = Ww[e - 114688];
  }
  dst[e] = (_Float16)v;
}

// ===========================================================================
// Kernel 1: TemporalLift  (conv1d k=5 -> relu -> mean_t -> fc -> relu) -> u f16
// one block per (b,n), 128 threads (thread = channel / output dim)
// ===========================================================================
__global__ void k_temporal(const float* __restrict__ spikes,
                           const float* __restrict__ conv_w,
                           const float* __restrict__ conv_b,
                           const float* __restrict__ fc_w,
                           const float* __restrict__ fc_b,
                           _Float16* __restrict__ u_out) {
  __shared__ float ssp[Tt + 4];
  __shared__ float sxm[128];
  const int node = blockIdx.x;
  const int tid  = threadIdx.x;
  const float* sp = spikes + (size_t)node * Tt;
  ssp[tid + 2]   = sp[tid];
  ssp[tid + 130] = sp[tid + 128];
  if (tid == 0) { ssp[0] = 0.f; ssp[1] = 0.f; ssp[Tt + 2] = 0.f; ssp[Tt + 3] = 0.f; }
  __syncthreads();

  const int h = tid;
  const float w0 = conv_w[h*5+0], w1 = conv_w[h*5+1], w2 = conv_w[h*5+2],
              w3 = conv_w[h*5+3], w4 = conv_w[h*5+4];
  const float bb = conv_b[h];
  float acc = 0.f;
  #pragma unroll 4
  for (int t = 0; t < Tt; ++t) {
    float s = fmaf(w0, ssp[t], fmaf(w1, ssp[t+1],
              fmaf(w2, ssp[t+2], fmaf(w3, ssp[t+3], fmaf(w4, ssp[t+4], bb)))));
    acc += fmaxf(s, 0.f);
  }
  sxm[h] = acc * (1.f / (float)Tt);
  __syncthreads();

  const int d = tid;
  float fa = fc_b[d];
  #pragma unroll 8
  for (int hh = 0; hh < Dd; ++hh) fa = fmaf(sxm[hh], fc_w[hh * Dd + d], fa);
  u_out[(size_t)node * Dd + d] = (_Float16)fmaxf(fa, 0.f);
}

// ===========================================================================
// Kernel 2: kNN on 1-D coords. One thread per node, insertion top-16.
// ===========================================================================
__global__ void k_knn(const float* __restrict__ coords, int* __restrict__ idx_out) {
  const int gid = blockIdx.x * blockDim.x + threadIdx.x; // 0..B*N-1
  const int b = gid >> 11;
  const int i = gid & (Nn - 1);
  const float* cb = coords + (size_t)b * Nn;
  const float xi = cb[i];
  float bd[Kk]; int bi[Kk];
  #pragma unroll
  for (int s = 0; s < Kk; ++s) { bd[s] = 3.0e38f; bi[s] = 0; }
  for (int j = 0; j < Nn; ++j) {
    const float df = xi - cb[j];
    const float d2 = df * df;
    if (d2 < bd[Kk - 1]) {
      bd[Kk - 1] = d2; bi[Kk - 1] = j;
      #pragma unroll
      for (int s = Kk - 1; s > 0; --s) {
        if (bd[s] < bd[s - 1]) {
          float td = bd[s]; bd[s] = bd[s-1]; bd[s-1] = td;
          int   ti = bi[s]; bi[s] = bi[s-1]; bi[s-1] = ti;
        }
      }
    }
  }
  #pragma unroll
  for (int s = 0; s < Kk; ++s) idx_out[(size_t)gid * Kk + s] = bi[s];
}

// ===========================================================================
// Kernel 3: one GraphKernelLayer. 256 threads = 8 waves, 1 node per wave.
//   h1 = relu(u_j @ W1uj  + [u_i@W1ui + x_i*w1x + b1]  + x_j (x) w1xj)
//   h2 = relu(h1 @ k2 + b2);  Kv = (h2 . k3 + b3)/4
//   u' = relu(u_i @ local_w + local_b + sum_k Kv_k u_jk + u_i)
// Dynamic LDS: sW(32K, TDM-staged) | sU(32K) | sH(32K) | aux
// ===========================================================================
__global__ void k_layer(const _Float16* __restrict__ u_in,
                        _Float16* __restrict__ u_out,
                        const int* __restrict__ idx,
                        const float* __restrict__ coords,
                        const float* __restrict__ localw,
                        const float* __restrict__ localb,
                        const float* __restrict__ k1w,
                        const float* __restrict__ k1b,
                        const _Float16* __restrict__ wuj16,  // f16 k1 u_j slice
                        const _Float16* __restrict__ k216,   // f16 k2
                        const float* __restrict__ k2b,
                        const float* __restrict__ k3w,
                        const float* __restrict__ k3b) {
  extern __shared__ char smem[];
  _Float16* sW  = (_Float16*)smem;             // 16384 halves, W1uj then k2
  _Float16* sU  = sW + 16384;                  // 8 waves * 16x128 (u_j)
  _Float16* sH  = sU + 16384;                  // 8 waves * 16x128 (h1)
  float* sAux   = (float*)(sH + 16384);        // 8 * 128  (v0 per node)
  float* sKv    = sAux + 8 * 128;              // 8 * 16
  float* sXj    = sKv + 8 * Kk;                // 8 * 16

  const int tid  = threadIdx.x;
  const int wave = tid >> 5;
  const int lane = tid & 31;
  const int node = blockIdx.x * 8 + wave;      // == b*N + i
  const int b    = node >> 11;
  const int i    = node & (Nn - 1);
  const size_t urow = (size_t)node * Dd;

  // ---- stage W1uj (f16) via TDM (wave 0) ----
  stage_weights(sW, wuj16, tid, wave, 0u);
  if (tid < 8 * Kk) sKv[tid] = 0.f;
  // warm L2 with k2 f16 weights for the restage later (global_prefetch_b8)
  __builtin_prefetch(k216 + tid * 64, 0, 1);

  // ---- gather u_j tile (16x128 f16) and x_j ----
  const int* idxp = idx + (size_t)node * Kk;
  #pragma unroll
  for (int q = 0; q < 8; ++q) {
    const int c = lane + q * 32;               // 256 chunks of 8 halves
    const int k = c >> 4, part = c & 15;
    const int nb = idxp[k];
    v8h v = *(const v8h*)(u_in + ((size_t)b * Nn + nb) * Dd + part * 8);
    *(v8h*)(sU + wave * 2048 + k * Dd + part * 8) = v;
  }
  if (lane < Kk) sXj[wave * Kk + lane] = coords[b * Nn + idxp[lane]];

  // ---- v0[d] = k1b[d] + x_i*k1w[0,d] + u_i @ k1w[2:130,:] (4 d's per lane) ----
  const float xi = coords[b * Nn + i];
  const int d0 = lane * 4;
  v4f accv;
  #pragma unroll
  for (int c = 0; c < 4; ++c) accv[c] = fmaf(xi, k1w[d0 + c], k1b[d0 + c]);
  for (int hh = 0; hh < Dd; ++hh) {
    const float uh = (float)u_in[urow + hh];
    v4f w = *(const v4f*)(k1w + (2 + hh) * Dd + d0);
    accv += uh * w;
  }
  #pragma unroll
  for (int c = 0; c < 4; ++c) sAux[wave * 128 + d0 + c] = accv[c];
  __syncthreads();

  const int mb = (lane >> 4) << 3;             // D-frag row base: 0 or 8

  // ---- GEMM1: h1 = relu(u_j @ W1uj + v0 + x_j (x) k1w[1,:]) ----
  {
    v8f acc[8] = {};
    #pragma unroll
    for (int ks = 0; ks < 4; ++ks) {
      v16h a = load_a_frag(sU + wave * 2048, ks * 32, lane);
      #pragma unroll
      for (int nt = 0; nt < 8; ++nt)
        acc[nt] = wmma_f16(a, load_b_frag(sW, ks * 32, nt, lane), acc[nt]);
    }
    #pragma unroll
    for (int nt = 0; nt < 8; ++nt) {
      const int n = nt * 16 + (lane & 15);
      const float add = sAux[wave * 128 + n];
      const float w1n = k1w[Dd + n];           // k1_w row 1 (x_j)
      #pragma unroll
      for (int r = 0; r < 8; ++r) {
        const int m = mb + r;
        const float val = acc[nt][r] + add + sXj[wave * Kk + m] * w1n;
        sH[wave * 2048 + m * Dd + n] = (_Float16)fmaxf(val, 0.f);
      }
    }
  }
  __syncthreads();

  // ---- restage k2 f16 weights into sW via TDM ----
  stage_weights(sW, k216, tid, wave, 0u);
  __syncthreads();

  // ---- GEMM2 + Kv reduction ----
  {
    v8f acc[8] = {};
    #pragma unroll
    for (int ks = 0; ks < 4; ++ks) {
      v16h a = load_a_frag(sH + wave * 2048, ks * 32, lane);
      #pragma unroll
      for (int nt = 0; nt < 8; ++nt)
        acc[nt] = wmma_f16(a, load_b_frag(sW, ks * 32, nt, lane), acc[nt]);
    }
    float kvloc[8] = {};
    #pragma unroll
    for (int nt = 0; nt < 8; ++nt) {
      const int n = nt * 16 + (lane & 15);
      const float b2n = k2b[n];
      const float k3n = k3w[n];
      #pragma unroll
      for (int r = 0; r < 8; ++r) {
        const float h2 = fmaxf(acc[nt][r] + b2n, 0.f);
        kvloc[r] = fmaf(h2, k3n, kvloc[r]);
      }
    }
    #pragma unroll
    for (int r = 0; r < 8; ++r) atomicAdd(&sKv[wave * Kk + mb + r], kvloc[r]);
  }
  __syncthreads();

  // ---- agg + local matvec + residual + relu -> u_out (f16) ----
  const float k3b0 = k3b[0];
  v4f out;
  #pragma unroll
  for (int c = 0; c < 4; ++c) out[c] = localb[d0 + c];
  #pragma unroll
  for (int k = 0; k < Kk; ++k) {
    const float kv = (sKv[wave * Kk + k] + k3b0) * 0.25f; // * 1/sqrt(K)
    #pragma unroll
    for (int c = 0; c < 4; ++c)
      out[c] = fmaf(kv, (float)sU[wave * 2048 + k * Dd + d0 + c], out[c]);
  }
  for (int hh = 0; hh < Dd; ++hh) {
    const float uh = (float)u_in[urow + hh];
    v4f w = *(const v4f*)(localw + hh * Dd + d0);
    out += uh * w;
  }
  #pragma unroll
  for (int c = 0; c < 4; ++c) {
    const float uic = (float)u_in[urow + d0 + c];
    u_out[urow + d0 + c] = (_Float16)fmaxf(out[c] + uic, 0.f);
  }
}

// ===========================================================================
// Kernel 4: v_l = u @ W_logits, v_w = u @ W_weight   (f16 out)
// 8 waves x 16 rows per block; f16 W TDM-staged one at a time in 32KB LDS.
// ===========================================================================
__global__ void k_project(const _Float16* __restrict__ u16,
                          const _Float16* __restrict__ wl16,
                          const _Float16* __restrict__ ww16,
                          _Float16* __restrict__ vl,
                          _Float16* __restrict__ vw) {
  extern __shared__ char smem[];
  _Float16* sW = (_Float16*)smem;              // 16384 halves
  const int tid = threadIdx.x, wave = tid >> 5, lane = tid & 31;
  const int rowbase = (blockIdx.x * 8 + wave) * 16;
  const _Float16* abase = u16 + (size_t)rowbase * Dd;

  v16h a[4];
  #pragma unroll
  for (int ks = 0; ks < 4; ++ks) a[ks] = load_a_frag(abase, ks * 32, lane);

  stage_weights(sW, wl16, tid, wave, 0u);
  __syncthreads();
  v8f accL[8] = {};
  #pragma unroll
  for (int ks = 0; ks < 4; ++ks)
    #pragma unroll
    for (int nt = 0; nt < 8; ++nt)
      accL[nt] = wmma_f16(a[ks], load_b_frag(sW, ks * 32, nt, lane), accL[nt]);
  __syncthreads();

  stage_weights(sW, ww16, tid, wave, 0u);
  __syncthreads();
  v8f accW[8] = {};
  #pragma unroll
  for (int ks = 0; ks < 4; ++ks)
    #pragma unroll
    for (int nt = 0; nt < 8; ++nt)
      accW[nt] = wmma_f16(a[ks], load_b_frag(sW, ks * 32, nt, lane), accW[nt]);

  const int mb = (lane >> 4) << 3;
  #pragma unroll
  for (int nt = 0; nt < 8; ++nt) {
    const int n = nt * 16 + (lane & 15);
    #pragma unroll
    for (int r = 0; r < 8; ++r) {
      const int row = rowbase + mb + r;
      vl[(size_t)row * Dd + n] = (_Float16)accL[nt][r];
      vw[(size_t)row * Dd + n] = (_Float16)accW[nt][r];
    }
  }
}

// ===========================================================================
// Kernel 5: G_l = v_l @ u^T, G_w = v_w @ u^T ; logits / sigmoid / softplus.
// 128x128 output tile per block (8 waves x 16 rows), u^T staged in LDS.
// ===========================================================================
__global__ void k_gram(const _Float16* __restrict__ vl,
                       const _Float16* __restrict__ vw,
                       const _Float16* __restrict__ u16,
                       float* __restrict__ outL,
                       float* __restrict__ outP,
                       float* __restrict__ outW) {
  extern __shared__ char smem[];
  _Float16* sUT = (_Float16*)smem;             // [d=128][j=128]
  const int tid = threadIdx.x, wave = tid >> 5, lane = tid & 31;
  const int b = blockIdx.z;
  const int jbase = blockIdx.x * 128;
  const int ibase = blockIdx.y * 128;

  // stage u tile transposed: sUT[d*128 + jloc] = u[jbase+jloc, d]
  for (int e = tid; e < 2048; e += 256) {
    const int jl = e >> 4, part = e & 15;
    v8h v = *(const v8h*)(u16 + ((size_t)b * Nn + jbase + jl) * Dd + part * 8);
    #pragma unroll
    for (int c = 0; c < 8; ++c) sUT[(part * 8 + c) * 128 + jl] = v[c];
  }
  __syncthreads();

  const int rglob = ibase + wave * 16;
  const _Float16* aLb = vl + ((size_t)b * Nn + rglob) * Dd;
  const _Float16* aWb = vw + ((size_t)b * Nn + rglob) * Dd;
  v8f accL[8] = {}, accW[8] = {};
  #pragma unroll
  for (int ks = 0; ks < 4; ++ks) {
    v16h al = load_a_frag(aLb, ks * 32, lane);
    v16h aw = load_a_frag(aWb, ks * 32, lane);
    #pragma unroll
    for (int nt = 0; nt < 8; ++nt) {
      v16h bf = load_b_frag(sUT, ks * 32, nt, lane);
      accL[nt] = wmma_f16(al, bf, accL[nt]);
      accW[nt] = wmma_f16(aw, bf, accW[nt]);
    }
  }

  const int mb = (lane >> 4) << 3;
  const size_t obase = (size_t)b * Nn * Nn;
  #pragma unroll
  for (int nt = 0; nt < 8; ++nt) {
    const int j = jbase + nt * 16 + (lane & 15);
    #pragma unroll
    for (int r = 0; r < 8; ++r) {
      const int i = rglob + mb + r;
      const size_t o = obase + (size_t)i * Nn + j;
      const float x = accL[nt][r];
      outL[o] = x;
      outP[o] = 1.f / (1.f + __expf(-x));
      const float w = accW[nt][r];
      outW[o] = (w > 20.f) ? w : log1pf(__expf(w));
    }
  }
}

} // anonymous namespace

// ===========================================================================
extern "C" void kernel_launch(void* const* d_in, const int* in_sizes, int n_in,
                              void* d_out, int out_size, void* d_ws, size_t ws_size,
                              hipStream_t stream) {
  (void)in_sizes; (void)n_in; (void)out_size; (void)ws_size;
  const float* spikes  = (const float*)d_in[0];
  const float* coords  = (const float*)d_in[1];
  const float* conv_w  = (const float*)d_in[2];
  const float* conv_b  = (const float*)d_in[3];
  const float* fc_w    = (const float*)d_in[4];
  const float* fc_b    = (const float*)d_in[5];
  const float* local_w = (const float*)d_in[6];
  const float* local_b = (const float*)d_in[7];
  const float* k1_w    = (const float*)d_in[8];
  const float* k1_b    = (const float*)d_in[9];
  const float* k2_w    = (const float*)d_in[10];
  const float* k2_b    = (const float*)d_in[11];
  const float* k3_w    = (const float*)d_in[12];
  const float* k3_b    = (const float*)d_in[13];
  const float* W_log   = (const float*)d_in[14];
  const float* W_wei   = (const float*)d_in[15];

  char* ws = (char*)d_ws;
  _Float16* uA   = (_Float16*)(ws);                             // 1 MB
  _Float16* uB   = (_Float16*)(ws + (1u << 20));                // 1 MB
  int*      idx  = (int*)     (ws + (2u << 20));                // 512 KB
  _Float16* vvl  = (_Float16*)(ws + (2u << 20) + (512u << 10)); // 512 KB
  _Float16* vvw  = (_Float16*)(ws + (3u << 20));                // 512 KB
  _Float16* wcvt = (_Float16*)(ws + (3u << 20) + (512u << 10)); // 256 KB f16 weights

  _Float16* wuj16 = wcvt;            // 3 * 16384
  _Float16* k216  = wcvt + 49152;    // 3 * 16384
  _Float16* wl16  = wcvt + 98304;    // 16384
  _Float16* ww16  = wcvt + 114688;   // 16384

  float* outL = (float*)d_out;
  float* outP = outL + (size_t)Bb * Nn * Nn;
  float* outW = outP + (size_t)Bb * Nn * Nn;

  constexpr int LDS_LAYER = 3 * 32768 + (8 * 128 + 8 * Kk + 8 * Kk) * 4; // 103424 B
  (void)hipFuncSetAttribute((const void*)k_layer,
                            hipFuncAttributeMaxDynamicSharedMemorySize, LDS_LAYER);

  k_convert<<<dim3(512), dim3(256), 0, stream>>>(k1_w, k2_w, W_log, W_wei, wcvt);
  k_temporal<<<dim3(Bb * Nn), dim3(128), 0, stream>>>(spikes, conv_w, conv_b, fc_w, fc_b, uA);
  k_knn<<<dim3((Bb * Nn) / 256), dim3(256), 0, stream>>>(coords, idx);

  const _Float16* uin = uA;
  _Float16* uout = uB;
  for (int l = 0; l < Ll; ++l) {
    k_layer<<<dim3((Bb * Nn) / 8), dim3(256), LDS_LAYER, stream>>>(
        uin, uout, idx, coords,
        local_w + (size_t)l * Dd * Dd, local_b + (size_t)l * Dd,
        k1_w + (size_t)l * 258 * Dd,   k1_b + (size_t)l * Dd,
        wuj16 + (size_t)l * 16384,     k216 + (size_t)l * 16384,
        k2_b + (size_t)l * Dd,
        k3_w + (size_t)l * Dd,         k3_b + l);
    const _Float16* t = uout; uout = (_Float16*)uin; uin = t;
  }
  // uin now points at the final u (f16)

  k_project<<<dim3((Bb * Nn) / 128), dim3(256), 32768, stream>>>(uin, wl16, ww16, vvl, vvw);
  k_gram<<<dim3(Nn / 128, Nn / 128, Bb), dim3(256), 32768, stream>>>(vvl, vvw, uin, outL, outP, outW);
}
